// Net_35089882808747
// MI455X (gfx1250) — matrix operand order repaired and checked by reading, not compiled, not run on hardware
//
#include <hip/hip_runtime.h>

typedef __attribute__((ext_vector_type(16))) _Float16 v16h;
typedef __attribute__((ext_vector_type(8)))  float    v8f;

#define NN 50000
#define EE 800000

// ---------------------------------------------------------------- utilities
__global__ void k_zero(float* __restrict__ p, long long n) {
    long long i = (long long)blockIdx.x * blockDim.x + threadIdx.x;
    long long st = (long long)gridDim.x * blockDim.x;
    for (; i < n; i += st) p[i] = 0.0f;
}

__global__ void k_degrees(const int* __restrict__ src, const int* __restrict__ dst,
                          float* __restrict__ deg_in, float* __restrict__ deg_out) {
    int e = blockIdx.x * blockDim.x + threadIdx.x;
    if (e < EE) {
        atomicAdd(&deg_in[dst[e]], 1.0f);
        atomicAdd(&deg_out[src[e]], 1.0f);
    }
}

__global__ void k_inv(const float* __restrict__ deg_in, const float* __restrict__ deg_out,
                      float* __restrict__ rcp_in, float* __restrict__ inv_in,
                      float* __restrict__ inv_out) {
    int i = blockIdx.x * blockDim.x + threadIdx.x;
    if (i < NN) {
        float di = fmaxf(deg_in[i], 1.0f);
        float dq = fmaxf(deg_out[i], 1.0f);
        rcp_in[i]  = 1.0f / di;
        inv_in[i]  = rsqrtf(di);
        inv_out[i] = rsqrtf(dq);
    }
}

__device__ __forceinline__ void pack8s(v16h& a, int base, float4 u, float4 v, float s) {
    a[base + 0] = (_Float16)(u.x * s); a[base + 1] = (_Float16)(u.y * s);
    a[base + 2] = (_Float16)(u.z * s); a[base + 3] = (_Float16)(u.w * s);
    a[base + 4] = (_Float16)(v.x * s); a[base + 5] = (_Float16)(v.y * s);
    a[base + 6] = (_Float16)(v.z * s); a[base + 7] = (_Float16)(v.w * s);
}

// WMMA A-fragment loader: 16 rows x 32 K from row-major f32 (contiguous k = 16*half + s).
__device__ __forceinline__ v16h load_a_frag(const float* arow, int kk, int half, float s) {
    int k0a = kk + half * 16, k0b = kk + half * 16 + 8;
    float4 u0 = *(const float4*)(arow + k0a);
    float4 u1 = *(const float4*)(arow + k0a + 4);
    float4 u2 = *(const float4*)(arow + k0b);
    float4 u3 = *(const float4*)(arow + k0b + 4);
    v16h a;
    pack8s(a, 0, u0, u1, s);
    pack8s(a, 8, u2, u3, s);
    return a;
}

// ------------------------------------------------- edge linear + scatter-sum
// One wave per 16-edge tile: eh = ef(16x32) @ We(32x32) + be, atomically
// scatter-added onto agg_e[dst].  We staged in LDS in B-fragment layout.
__global__ void k_edge_wmma(const float* __restrict__ ef, const float* __restrict__ We,
                            const float* __restrict__ be, const int* __restrict__ dst,
                            float* __restrict__ agg_e) {
    __shared__ _Float16 sB[2 * 32 * 16];  // [tn][lane][i] : 2KB

    // cooperative stage: coalesced read of We (32x32), scatter to fragment layout
    for (int widx = threadIdx.x; widx < 32 * 32; widx += blockDim.x) {
        int k = widx >> 5, n = widx & 31;
        int tnl = n >> 4, nl = n & 15;
        int kh = k >> 4, i = k & 15;  // K=32: half = k/16, elem = k%16
        sB[((tnl * 32) + (kh * 16 + nl)) * 16 + i] = (_Float16)We[widx];
    }
    __syncthreads();

    int wave = (blockIdx.x * blockDim.x + threadIdx.x) >> 5;
    if (wave >= EE / 16) return;
    int lane = threadIdx.x & 31;
    int half = lane >> 4;
    int mr   = lane & 15;

    const float* arow = ef + (size_t)(wave * 16 + mr) * 32;
    v16h a = load_a_frag(arow, 0, half, 1.0f);

#pragma unroll
    for (int tn = 0; tn < 2; ++tn) {
        int n = tn * 16 + mr;
        v16h b = *(const v16h*)&sB[(tn * 32 + lane) * 16];
        v8f c = {};
        c = __builtin_amdgcn_wmma_f32_16x16x32_f16(false, a, false, b, (short)0, c,
                                                   false, false);
        float bias = be[n];
#pragma unroll
        for (int r = 0; r < 8; ++r) {
            int e = wave * 16 + r + half * 8;
            atomicAdd(&agg_e[(size_t)dst[e] * 32 + n], c[r] + bias);
        }
    }
}

// edge_h = relu(agg_e / deg_in) into h0 columns [0,32)
__global__ void k_edge_h(const float* __restrict__ agg_e, const float* __restrict__ rcp_in,
                         float* __restrict__ h0) {
    int i = blockIdx.x * blockDim.x + threadIdx.x;
    if (i < NN * 32) {
        int node = i >> 5, col = i & 31;
        h0[(size_t)node * 128 + col] = fmaxf(agg_e[i] * rcp_in[node], 0.0f);
    }
}

// ------------------------------------------------------- strip-mined WMMA GEMM
// Out[m, col_off+n] = act( (A[m,:]*scale[m]) @ W[:,n] + bias[n] )
// Each wave owns one 16-row strip and TPP=NOUT/16/NPASS column tiles per pass.
// W is converted to f16 and staged in LDS in WMMA B-fragment layout per pass.
template <int K, int NOUT, int NPASS>
__global__ void k_gemm(const float* __restrict__ A, int lda,
                       const float* __restrict__ scale,
                       const float* __restrict__ W, const float* __restrict__ bias,
                       float* __restrict__ Out, int out_stride, int out_col_off,
                       int M, int relu) {
    constexpr int NKS = K / 32;            // k-steps
    constexpr int NTN = NOUT / 16;         // column tiles
    constexpr int TPP = NTN / NPASS;       // column tiles per pass
    __shared__ _Float16 sB[NKS * TPP * 32 * 16];

    int lane = threadIdx.x & 31;
    int half = lane >> 4;
    int mr   = lane & 15;
    int strip = (blockIdx.x * blockDim.x + threadIdx.x) >> 5;
    bool active = strip < (M >> 4);        // wave-uniform

    int m = strip * 16 + mr;
    const float* arow = A + (size_t)(active ? m : 0) * lda;
    float s = (active && scale) ? scale[m] : 1.0f;

    for (int pass = 0; pass < NPASS; ++pass) {
        int n0 = pass * TPP * 16;
        __syncthreads();   // previous-pass LDS reads done
        // cooperative stage: coalesced over output columns of this pass
        for (int widx = threadIdx.x; widx < K * TPP * 16; widx += blockDim.x) {
            int k  = widx / (TPP * 16);
            int nc = widx % (TPP * 16);
            int tnl = nc >> 4, nl = nc & 15;
            int ks = k >> 5, kh = (k >> 4) & 1, i = k & 15;
            sB[(((ks * TPP + tnl) * 32) + (kh * 16 + nl)) * 16 + i] =
                (_Float16)W[(size_t)k * NOUT + n0 + nc];
        }
        __syncthreads();

        if (active) {
            v8f acc[TPP];
#pragma unroll
            for (int t = 0; t < TPP; ++t) acc[t] = (v8f){};
#pragma unroll
            for (int ks = 0; ks < NKS; ++ks) {
                v16h a = load_a_frag(arow, ks * 32, half, s);
#pragma unroll
                for (int t = 0; t < TPP; ++t) {
                    v16h b = *(const v16h*)&sB[((ks * TPP + t) * 32 + lane) * 16];
                    acc[t] = __builtin_amdgcn_wmma_f32_16x16x32_f16(
                        false, a, false, b, (short)0, acc[t], false, false);
                }
            }
#pragma unroll
            for (int t = 0; t < TPP; ++t) {
                int n = n0 + t * 16 + mr;
                float bb = bias[n];
#pragma unroll
                for (int r = 0; r < 8; ++r) {
                    int mm = strip * 16 + r + half * 8;
                    float v = acc[t][r] + bb;
                    if (relu) v = fmaxf(v, 0.0f);
                    Out[(size_t)mm * out_stride + out_col_off + n] = v;
                }
            }
        }
    }
}

// ------------------------------------------- conv message gather/scatter-add
// agg[dst] += h[src] * inv_out[src] ; vectorized float4 gather, f32 atomics.
__global__ void k_scatter(const float* __restrict__ h, const float* __restrict__ inv_out,
                          const int* __restrict__ src, const int* __restrict__ dst,
                          float* __restrict__ agg, int d) {
    int dq = d >> 2;
    long long total = (long long)EE * dq;
    long long idx = (long long)blockIdx.x * blockDim.x + threadIdx.x;
    long long st  = (long long)gridDim.x * blockDim.x;
    for (; idx < total; idx += st) {
        int e  = (int)(idx / dq);
        int c4 = (int)(idx % dq) * 4;
        int sv = src[e];
        float sc = inv_out[sv];
        float4 v = *(const float4*)(h + (size_t)sv * d + c4);
        float* out = agg + (size_t)dst[e] * d + c4;
        atomicAdd(out + 0, v.x * sc);
        atomicAdd(out + 1, v.y * sc);
        atomicAdd(out + 2, v.z * sc);
        atomicAdd(out + 3, v.w * sc);
    }
}

// --------------------------------------------------------------- final head
__global__ void k_out(const float* __restrict__ h, const float* __restrict__ Wo,
                      const float* __restrict__ bo, float* __restrict__ out) {
    int n = blockIdx.x * blockDim.x + threadIdx.x;
    if (n >= NN) return;
    const float* row = h + (size_t)n * 192;
    float a0 = bo[0], a1 = bo[1];
#pragma unroll 4
    for (int k = 0; k < 192; ++k) {
        float a = row[k];
        a0 += a * Wo[k * 2 + 0];
        a1 += a * Wo[k * 2 + 1];
    }
    out[n * 2 + 0] = a0;
    out[n * 2 + 1] = a1;
}

extern "C" void kernel_launch(void* const* d_in, const int* in_sizes, int n_in,
                              void* d_out, int out_size, void* d_ws, size_t ws_size,
                              hipStream_t stream) {
    (void)in_sizes; (void)n_in; (void)out_size; (void)ws_size;
    const float* node_feats = (const float*)d_in[0];   // N x 64
    const float* edge_feats = (const float*)d_in[1];   // E x 32
    const float* Wn  = (const float*)d_in[2];          // 64 x 96
    const float* bn  = (const float*)d_in[3];
    const float* We  = (const float*)d_in[4];          // 32 x 32
    const float* be  = (const float*)d_in[5];
    const float* Wc0 = (const float*)d_in[6];          // 128 x 192
    const float* bc0 = (const float*)d_in[7];
    const float* Wc1 = (const float*)d_in[8];          // 192 x 192
    const float* bc1 = (const float*)d_in[9];
    const float* Wl0 = (const float*)d_in[10];         // 192 x 192
    const float* bl0 = (const float*)d_in[11];
    const float* Wo  = (const float*)d_in[12];         // 192 x 2
    const float* bo  = (const float*)d_in[13];
    const int*   src = (const int*)d_in[14];
    const int*   dst = (const int*)d_in[15];
    float* out = (float*)d_out;

    // workspace layout (floats)
    float* w = (float*)d_ws;
    size_t o = 0;
    float* deg_in  = w + o; o += NN;
    float* deg_out = w + o; o += NN;
    float* rcp_in  = w + o; o += NN;
    float* inv_in  = w + o; o += NN;
    float* inv_out = w + o; o += NN;
    float* agg_e   = w + o; o += (size_t)NN * 32;
    float* H0      = w + o; o += (size_t)NN * 192;
    float* H1      = w + o; o += (size_t)NN * 192;
    float* AGG     = w + o; o += (size_t)NN * 192;

    const int B = 256;
    const int strips = NN / 16;                      // 3125 row strips
    const int gemm_blocks = (strips + 7) / 8;        // 8 waves per block

    // degrees + norms
    k_zero<<<1024, B, 0, stream>>>(deg_in, 2LL * NN);  // deg_in + deg_out
    k_zero<<<4096, B, 0, stream>>>(agg_e, 32LL * NN);
    k_degrees<<<(EE + B - 1) / B, B, 0, stream>>>(src, dst, deg_in, deg_out);
    k_inv<<<(NN + B - 1) / B, B, 0, stream>>>(deg_in, deg_out, rcp_in, inv_in, inv_out);

    // edge layer: WMMA per-edge linear + scatter-sum, then mean+relu into H0[:,0:32]
    k_edge_wmma<<<(EE / 16 + 7) / 8, B, 0, stream>>>(edge_feats, We, be, dst, agg_e);
    k_edge_h<<<(NN * 32 + B - 1) / B, B, 0, stream>>>(agg_e, rcp_in, H0);

    // node embedding: relu(node_feats @ Wn + bn) into H0[:,32:128]
    k_gemm<64, 96, 1><<<gemm_blocks, B, 0, stream>>>(node_feats, 64, nullptr, Wn, bn,
                                                     H0, 128, 32, NN, 1);

    // conv0: agg = scatter(D_out^-1/2 h0) ; H1 = relu(D_in^-1/2 agg @ Wc0 + bc0)
    k_zero<<<4096, B, 0, stream>>>(AGG, 128LL * NN);
    k_scatter<<<8192, B, 0, stream>>>(H0, inv_out, src, dst, AGG, 128);
    k_gemm<128, 192, 2><<<gemm_blocks, B, 0, stream>>>(AGG, 128, inv_in, Wc0, bc0,
                                                       H1, 192, 0, NN, 1);

    // conv1
    k_zero<<<4096, B, 0, stream>>>(AGG, 192LL * NN);
    k_scatter<<<8192, B, 0, stream>>>(H1, inv_out, src, dst, AGG, 192);
    k_gemm<192, 192, 2><<<gemm_blocks, B, 0, stream>>>(AGG, 192, inv_in, Wc1, bc1,
                                                       H0, 192, 0, NN, 1);

    // hidden linear
    k_gemm<192, 192, 2><<<gemm_blocks, B, 0, stream>>>(H0, 192, nullptr, Wl0, bl0,
                                                       H1, 192, 0, NN, 1);

    // classifier head
    k_out<<<(NN + B - 1) / B, B, 0, stream>>>(H1, Wo, bo, out);
}
